// SymmetricContraction_54597624266897
// MI455X (gfx1250) — matrix operand order repaired and checked by reading, not compiled, not run on hardware
//
#include <hip/hip_runtime.h>
#include <hip/hip_bf16.h>

// MACE symmetric contraction (corr. order 3) for MI455X / gfx1250, wave32.
//
// Formulation: per (n,c) column, out = xx^T * (Aext @ v) + U1-term, where
//   Aext[(ab),(kk)] : kk<368 -> U3[a,b,i,k] (ik flattened), kk in [368,372) -> U2[a,b,k], else 0
//   v[(kk)]         : kk<368 -> x[n,i,c]*w3[n,k,c],        kk in [368,372) -> w2[n,k,c],  else 0
//   xx[(ab)]        : x[n,a,c]*x[n,b,c]
// Big GEMM (M=256, K=384, Ncols=N*C=262144) runs on v_wmma_f32_16x16x32_f16
// (f16 operands, f32 accumulate). Aext is staged in LDS as f16 once per WG.
//
// NOTE: the M-tile order is rotated by `tile` ( m = (mi+tile)&15 ) so the
// per-M A-fragment LDS loads are NOT loop-invariant w.r.t. the tile loop;
// otherwise LICM hoists all 192 fragments and regalloc spills them to
// scratch (observed in round 1 disasm).
//
// WMMA wave32 operand layouts assumed per CDNA5 ISA 7.12.2:
//   A (16x32 f16): lane L -> row M=L%16; two contiguous 8-half chunks per lane:
//                  K = ks*32 + 8*hi + [0..7] and +16
//   B (32x16 f16): lane L -> col N=L%16; element e -> K = ks*32 + 8*hi + e, +8 if e>=8 (same order as A)
//   D (16x16 f32): lane L -> col N=L%16; element r -> row M = r + 8*hi

typedef __attribute__((ext_vector_type(16))) _Float16 v16h;
typedef __attribute__((ext_vector_type(8)))  _Float16 v8h;
typedef __attribute__((ext_vector_type(8)))  float    v8f;

#define NL   16
#define CCH  128
#define P3   23
#define P2   4
#define KIK  (NL * P3)        // 368
#define KEXT 384              // padded K (12 x 32)
#define KPAD 392              // LDS row stride in halves (bank-conflict pad)
#define MROW 256              // ab pairs
#define LDS_BYTES (MROW * KPAD * 2)

__device__ __forceinline__ float vext_val(int kk, const float* xc,
                                          const float* w3c, const float* w2c) {
  // kk is a compile-time constant after full unroll -> /23 and %23 fold away
  // and xc/w3c stay in registers.
  if (kk < KIK) {
    const int i  = kk / P3;
    const int k3 = kk - i * P3;
    return xc[i] * w3c[k3];
  }
  if (kk < KIK + P2) return w2c[kk - KIK];
  return 0.0f;
}

__global__ __launch_bounds__(512, 1)
void SymmetricContraction_54597624266897_kernel(
    const float* __restrict__ x,          // [N,16,128]
    const int*   __restrict__ atom_types, // [N]
    const float* __restrict__ U3,         // [16,16,16,23]
    const float* __restrict__ U2,         // [16,16,4]
    const float* __restrict__ U1,         // [16,1]
    const float* __restrict__ W3,         // [E,23,128]
    const float* __restrict__ W2,         // [E,4,128]
    const float* __restrict__ W1,         // [E,1,128]
    float*       __restrict__ out,        // [N,128]
    int num_tiles) {                      // N * (128/16)

  extern __shared__ _Float16 sU[];        // Aext in f16: [256][392]

  // ---- Cooperative stage: Aext (U3 ++ U2 ++ zeros) f32 -> f16 into LDS ----
  for (int idx = threadIdx.x; idx < MROW * KEXT; idx += blockDim.x) {
    const int ab = idx / KEXT;
    const int kk = idx - ab * KEXT;
    float v;
    if (kk < KIK)            v = U3[ab * KIK + kk];
    else if (kk < KIK + P2)  v = U2[ab * P2 + (kk - KIK)];
    else                     v = 0.0f;
    sU[ab * KPAD + kk] = (_Float16)v;
  }
  __syncthreads();

  const int lane  = threadIdx.x & 31;
  const int half  = lane & 15;            // column within tile / row within M-tile
  const bool hi   = lane >= 16;
  const int hi8   = hi ? 8 : 0;

  const int wavesPerBlk = blockDim.x >> 5;
  const int gw = blockIdx.x * wavesPerBlk + (threadIdx.x >> 5);
  const int nw = gridDim.x * wavesPerBlk;

  for (int tile = gw; tile < num_tiles; tile += nw) {
    const int n  = tile >> 3;             // atom
    const int c  = ((tile & 7) << 4) + half;  // channel for this lane's column
    const int t  = atom_types[n];

    // ---- Per-column operand gather (coalesced across the 16 columns) ----
    float xcol[NL];
#pragma unroll
    for (int i = 0; i < NL; ++i) xcol[i] = x[(n * NL + i) * CCH + c];
    float w3col[P3];
#pragma unroll
    for (int k = 0; k < P3; ++k) w3col[k] = W3[(t * P3 + k) * CCH + c];
    float w2col[P2];
#pragma unroll
    for (int k = 0; k < P2; ++k) w2col[k] = W2[(t * P2 + k) * CCH + c];
    const float w1c = W1[t * CCH + c];

    float u1dot = 0.0f;
#pragma unroll
    for (int a = 0; a < NL; ++a) u1dot = fmaf(U1[a], xcol[a], u1dot);
    const float u1term = w1c * u1dot;

    // x[b] for this lane's D rows: b = r + 8*hi
    float xb[8];
#pragma unroll
    for (int r = 0; r < 8; ++r) xb[r] = hi ? xcol[r + 8] : xcol[r];

    // ---- Build all 12 B fragments (V column) in registers, f16 ----
    v16h bfrag[12];
#pragma unroll
    for (int ks = 0; ks < 12; ++ks) {
      v16h bf;
#pragma unroll
      for (int e = 0; e < 16; ++e) {
        const int kk0 = ks * 32 + (e < 8 ? e : e + 8);  // lanes 0-15
        const int kk1 = kk0 + 8;           // lanes 16-31
        const float v0 = vext_val(kk0, xcol, w3col, w2col);
        const float v1 = vext_val(kk1, xcol, w3col, w2col);
        bf[e] = (_Float16)(hi ? v1 : v0);
      }
      bfrag[ks] = bf;
    }

    // ---- 16 M-tiles x 12 K-steps of WMMA, folding epilogue per M-tile ----
    // M order rotated by tile to defeat LICM (see header comment).
    float s = 0.0f;
#pragma unroll
    for (int mi = 0; mi < 16; ++mi) {
      const int m = (mi + tile) & 15;
      const _Float16* rowp = sU + (m * 16 + half) * KPAD;
      v8f acc = {};
#pragma unroll
      for (int ks = 0; ks < 12; ++ks) {
        const int base = ks * 32 + hi8;
        const v8h alo = *(const v8h*)(rowp + base);
        const v8h ahi = *(const v8h*)(rowp + base + 16);
        const v16h af = __builtin_shufflevector(alo, ahi,
            0, 1, 2, 3, 4, 5, 6, 7, 8, 9, 10, 11, 12, 13, 14, 15);
        acc = __builtin_amdgcn_wmma_f32_16x16x32_f16(
            false, af, false, bfrag[ks], (short)0, acc, false, false);
      }
      // acc[r] = c2[ab = m*16 + r + 8*hi] for this lane's column; a=m, b=r+8*hi.
      // m is runtime (rotated), so re-load x[n,m,c] from global (L0-hot,
      // coalesced) instead of dynamically indexing the xcol registers.
      const float xm = x[(n * NL + m) * CCH + c];
#pragma unroll
      for (int r = 0; r < 8; ++r) s = fmaf(acc[r] * xb[r], xm, s);
    }

    // Lanes L and L+16 hold the same column; merge their ab-halves.
    const float tot = s + __shfl_xor(s, 16, 32);
    if (!hi) out[n * CCH + c] = tot + u1term;
  }
}

extern "C" void kernel_launch(void* const* d_in, const int* in_sizes, int n_in,
                              void* d_out, int out_size, void* d_ws, size_t ws_size,
                              hipStream_t stream) {
  const float* x  = (const float*)d_in[0];
  const int*   at = (const int*)  d_in[1];
  const float* U3 = (const float*)d_in[2];
  const float* U2 = (const float*)d_in[3];
  const float* U1 = (const float*)d_in[4];
  const float* W3 = (const float*)d_in[5];
  const float* W2 = (const float*)d_in[6];
  const float* W1 = (const float*)d_in[7];
  float* out = (float*)d_out;

  const int N = in_sizes[1];              // atom count
  const int num_tiles = N * (CCH / 16);   // 16-channel column tiles

  dim3 grid(256);
  dim3 block(512);                        // 16 wave32s per WG
  SymmetricContraction_54597624266897_kernel<<<grid, block, LDS_BYTES, stream>>>(
      x, at, U3, U2, U1, W3, W2, W1, out, num_tiles);
}
